// TransformerBlock3D_18562848653468
// MI455X (gfx1250) — matrix operand order repaired and checked by reading, not compiled, not run on hardware
//
#include <hip/hip_runtime.h>
#include <math.h>

#define N_TOK   4096
#define C_DIM   256
#define FF_DIM  1024
#define N_HEADS 8
#define HEAD_DIM 32

typedef __attribute__((ext_vector_type(16))) __bf16 v16bf;
typedef __attribute__((ext_vector_type(8)))  __bf16 bf16x8;
typedef __attribute__((ext_vector_type(8)))  float  v8f;
typedef __attribute__((ext_vector_type(4)))  int    i32x4;

union BF16x16 { v16bf v; __bf16 e[16]; };

// ---- CDNA5 async global->LDS path (ASYNCcnt), guarded so build cannot break
#if __has_builtin(__builtin_amdgcn_global_load_async_to_lds_b128)
#define HAVE_ASYNC_LDS 1
// builtin signature (from clang diagnostic): (int4 AS1*, int4 AS3*, imm, cpol)
typedef __attribute__((address_space(1))) i32x4* as1_i4p;
typedef __attribute__((address_space(3))) i32x4* as3_i4p;
#else
#define HAVE_ASYNC_LDS 0
#endif

static __device__ inline void wait_async_lds() {
#if __has_builtin(__builtin_amdgcn_s_wait_asynccnt)
  __builtin_amdgcn_s_wait_asynccnt(0);
#endif
}

// copy 32 contiguous bytes (16 bf16) global -> LDS for this lane
static __device__ inline void stage32(const __bf16* src, __bf16* dst) {
#if HAVE_ASYNC_LDS
  __builtin_amdgcn_global_load_async_to_lds_b128((as1_i4p)(src),     (as3_i4p)(dst),     0, 0);
  __builtin_amdgcn_global_load_async_to_lds_b128((as1_i4p)(src + 8), (as3_i4p)(dst + 8), 0, 0);
#else
  *(bf16x8*)dst       = *(const bf16x8*)src;
  *(bf16x8*)(dst + 8) = *(const bf16x8*)(src + 8);
#endif
}

static __device__ inline v8f wmma_bf16f32(v16bf a, v16bf b, v8f c) {
  return __builtin_amdgcn_wmma_f32_16x16x32_bf16(false, a, false, b, (short)0, c, false, false);
}

static __device__ inline v16bf join8(bf16x8 lo, bf16x8 hi) {
  return __builtin_shufflevector(lo, hi, 0,1,2,3,4,5,6,7,8,9,10,11,12,13,14,15);
}

static __device__ inline float half16_max(float x) {
  x = fmaxf(x, __shfl_xor(x, 1, 32));
  x = fmaxf(x, __shfl_xor(x, 2, 32));
  x = fmaxf(x, __shfl_xor(x, 4, 32));
  x = fmaxf(x, __shfl_xor(x, 8, 32));
  return x;
}
static __device__ inline float half16_sum(float x) {
  x += __shfl_xor(x, 1, 32);
  x += __shfl_xor(x, 2, 32);
  x += __shfl_xor(x, 4, 32);
  x += __shfl_xor(x, 8, 32);
  return x;
}

// ---------------- weight convert + transpose: fp32 [r][c] -> bf16 [c][r] ----
__global__ __launch_bounds__(256)
void convert_transpose_kernel(const float* __restrict__ in, __bf16* __restrict__ out,
                              int rows, int cols) {
  int i = blockIdx.x * 256 + threadIdx.x;
  if (i >= rows * cols) return;
  int r = i / cols, c = i - r * cols;
  out[(size_t)c * rows + r] = (__bf16)in[i];
}

// ---------------- GroupNorm over channel-major x, emit token-major f32+bf16 -
__global__ __launch_bounds__(256)
void groupnorm1_kernel(const float* __restrict__ x, const float* __restrict__ gw,
                       const float* __restrict__ gb, float* __restrict__ t,
                       __bf16* __restrict__ tbf) {
  __shared__ float ssum[256], ssq[256];
  const int g = blockIdx.x, tid = threadIdx.x;
  const float* xg = x + (size_t)g * 8 * N_TOK;   // group region is contiguous
  float sum = 0.f, sq = 0.f;
  for (int i = tid; i < 8 * N_TOK; i += 256) { float v = xg[i]; sum += v; sq += v * v; }
  ssum[tid] = sum; ssq[tid] = sq;
  __syncthreads();
  for (int o = 128; o > 0; o >>= 1) {
    if (tid < o) { ssum[tid] += ssum[tid + o]; ssq[tid] += ssq[tid + o]; }
    __syncthreads();
  }
  const float inv_n = 1.0f / (8.0f * N_TOK);
  const float mu = ssum[0] * inv_n;
  const float rstd = rsqrtf(ssq[0] * inv_n - mu * mu + 1e-5f);
  for (int i = tid; i < 8 * N_TOK; i += 256) {
    int ch = g * 8 + (i >> 12);
    int s  = i & (N_TOK - 1);
    float val = (xg[i] - mu) * rstd * gw[ch] + gb[ch];
    t  [(size_t)s * C_DIM + ch] = val;
    tbf[(size_t)s * C_DIM + ch] = (__bf16)val;
  }
}

// ---------------- GroupNorm over token-major input -------------------------
__global__ __launch_bounds__(256)
void groupnorm2_kernel(const float* __restrict__ xin, const float* __restrict__ gw,
                       const float* __restrict__ gb, float* __restrict__ t2,
                       __bf16* __restrict__ t2bf) {
  __shared__ float ssum[256], ssq[256];
  const int g = blockIdx.x, tid = threadIdx.x;
  float sum = 0.f, sq = 0.f;
  for (int n = tid; n < N_TOK; n += 256) {
    const float* p = xin + (size_t)n * C_DIM + g * 8;
#pragma unroll
    for (int j = 0; j < 8; ++j) { float v = p[j]; sum += v; sq += v * v; }
  }
  ssum[tid] = sum; ssq[tid] = sq;
  __syncthreads();
  for (int o = 128; o > 0; o >>= 1) {
    if (tid < o) { ssum[tid] += ssum[tid + o]; ssq[tid] += ssq[tid + o]; }
    __syncthreads();
  }
  const float inv_n = 1.0f / (8.0f * N_TOK);
  const float mu = ssum[0] * inv_n;
  const float rstd = rsqrtf(ssq[0] * inv_n - mu * mu + 1e-5f);
  for (int n = tid; n < N_TOK; n += 256) {
    const float* p = xin + (size_t)n * C_DIM + g * 8;
#pragma unroll
    for (int j = 0; j < 8; ++j) {
      float val = (p[j] - mu) * rstd * gw[g * 8 + j] + gb[g * 8 + j];
      t2  [(size_t)n * C_DIM + g * 8 + j] = val;
      t2bf[(size_t)n * C_DIM + g * 8 + j] = (__bf16)val;
    }
  }
}

// ---------------- generic WMMA GEMM: C = A[MxK] * Bt[NxK]^T + bias ----------
// B tile (64 cols x 32 K) staged in double-buffered LDS via async copies,
// shared by all 4 waves; A fragments stream from global per wave.
enum GemmMode { GEMM_BIAS_BF16 = 0, GEMM_BIAS_BF16_T = 1, GEMM_BIAS_RES_F32 = 2,
                GEMM_BIAS_GELU_BF16 = 3, GEMM_BIAS_RES_F32_T = 4 };

#define BPAD 40   // row stride in halfwords (80B) -> conflict-free b128 reads

__global__ __launch_bounds__(128)
void wmma_gemm_kernel(const __bf16* __restrict__ A, const __bf16* __restrict__ Bt,
                      const float* __restrict__ bias, const float* __restrict__ res,
                      void* __restrict__ out, int M, int N, int K, int mode) {
  __shared__ __bf16 lds_b[2][64][BPAD];
  const int tid  = threadIdx.x;
  const int lane = tid & 31, wave = tid >> 5;
  const int hh = lane >> 4, nl = lane & 15;
  const int mbase = blockIdx.x * 64 + wave * 16;
  const int nbase = blockIdx.y * 64;

  // staging role: thread copies 32B of column (nbase+ccol), K-segment cseg
  const int ccol = tid >> 1, cseg = tid & 1;
  const __bf16* bsrc = Bt + (size_t)(nbase + ccol) * K + cseg * 16;

  // A fragment base for this lane (CDNA5 A-layout: chunk base = (lane/16)*8)
  const __bf16* apt = A + (size_t)(mbase + nl) * K + hh * 8;

  stage32(bsrc, &lds_b[0][ccol][cseg * 16]);

  v8f acc[4] = {};
  const int nsteps = K >> 5;
  for (int s = 0; s < nsteps; ++s) {
    const int pb = s & 1;
    wait_async_lds();
    __syncthreads();                       // tile pb ready for everyone
    if (s + 1 < nsteps)
      stage32(bsrc + (s + 1) * 32, &lds_b[pb ^ 1][ccol][cseg * 16]);

    bf16x8 alo = *(const bf16x8*)(apt + s * 32);
    bf16x8 ahi = *(const bf16x8*)(apt + s * 32 + 16);
    v16bf a = join8(alo, ahi);

    v16bf bfr[4];                          // issue all 8 ds_load_b128 first,
#pragma unroll                             // then back-to-back WMMAs
    for (int f = 0; f < 4; ++f) {
      const __bf16* lp = &lds_b[pb][nl + 16 * f][0] + hh * 8;
      bfr[f] = join8(*(const bf16x8*)lp, *(const bf16x8*)(lp + 16));
    }
#pragma unroll
    for (int f = 0; f < 4; ++f)
      acc[f] = wmma_bf16f32(a, bfr[f], acc[f]);

    __syncthreads();                       // all reads of pb done
  }

#pragma unroll
  for (int f = 0; f < 4; ++f) {
    const int n = nbase + 16 * f + nl;
    const float bv = bias[n];
#pragma unroll
    for (int v = 0; v < 8; ++v) {
      const int m = mbase + 8 * hh + v;    // C layout: row = v + 8*(lane/16)
      float val = acc[f][v] + bv;
      if (mode == GEMM_BIAS_BF16) {
        ((__bf16*)out)[(size_t)m * N + n] = (__bf16)val;
      } else if (mode == GEMM_BIAS_BF16_T) {
        ((__bf16*)out)[(size_t)n * M + m] = (__bf16)val;
      } else if (mode == GEMM_BIAS_RES_F32) {
        ((float*)out)[(size_t)m * N + n] = val + res[(size_t)m * N + n];
      } else if (mode == GEMM_BIAS_GELU_BF16) {
        float g = 0.5f * val * (1.0f + erff(val * 0.70710678118654752f));
        ((__bf16*)out)[(size_t)m * N + n] = (__bf16)g;
      } else { // GEMM_BIAS_RES_F32_T: fused transpose back to channel-major
        ((float*)out)[(size_t)n * M + m] = val + res[(size_t)m * N + n];
      }
    }
  }
}

// ---------------- flash attention: 1 wave = 1 head x 16 query rows ----------
// K fragments for chunk jb+32 load during softmax of chunk jb; V fragments
// issue at the top of the iteration so the P@V WMMAs never wait on memory.
__global__ __launch_bounds__(32)
void flash_attn_kernel(const __bf16* __restrict__ Q, const __bf16* __restrict__ Km,
                       const __bf16* __restrict__ Vt, __bf16* __restrict__ O) {
  __shared__ float pbuf[16][36];
  const int lane = threadIdx.x & 31;
  const int head = blockIdx.y;
  const int qb   = blockIdx.x * 16;
  const int hh = lane >> 4, nl = lane & 15;
  const float sm_scale = 0.17677669529663687f;  // 1/sqrt(32)

  // Q A-fragment, held in registers for the whole sweep
  const __bf16* qp = Q + (size_t)(qb + nl) * C_DIM + head * HEAD_DIM + hh * 16;
  v16bf a_q = join8(*(const bf16x8*)qp, *(const bf16x8*)(qp + 8));

  // hoisted fragment base pointers
  const __bf16* kp0 = Km + (size_t)nl * C_DIM + head * HEAD_DIM + hh * 16;
  const __bf16* kp1 = kp0 + (size_t)16 * C_DIM;
  const __bf16* vp0 = Vt + (size_t)(head * HEAD_DIM + nl) * N_TOK + hh * 16;
  const __bf16* vp1 = vp0 + (size_t)16 * N_TOK;

  v8f o0 = {}, o1 = {};
  float mrow[8], lrow[8];
#pragma unroll
  for (int v = 0; v < 8; ++v) { mrow[v] = -3.0e38f; lrow[v] = 0.0f; }

  v16bf bk0 = *(const v16bf*)kp0;                 // pipeline prologue
  v16bf bk1 = *(const v16bf*)kp1;

  for (int jb = 0; jb < N_TOK; jb += 32) {
    v16bf bv0 = *(const v16bf*)(vp0 + jb);        // in flight across softmax
    v16bf bv1 = *(const v16bf*)(vp1 + jb);
    v8f z = {};
    v8f s0 = wmma_bf16f32(a_q, bk0, z);
    v8f s1 = wmma_bf16f32(a_q, bk1, z);
    if (jb + 32 < N_TOK) {                        // prefetch next K chunk
      bk0 = *(const v16bf*)(kp0 + (size_t)(jb + 32) * C_DIM);
      bk1 = *(const v16bf*)(kp1 + (size_t)(jb + 32) * C_DIM);
    }

#pragma unroll
    for (int v = 0; v < 8; ++v) {                 // online softmax, row v+8*hh
      float e0 = s0[v] * sm_scale;
      float e1 = s1[v] * sm_scale;
      float mn = fmaxf(mrow[v], half16_max(fmaxf(e0, e1)));
      float sc = __expf(mrow[v] - mn);
      float p0 = __expf(e0 - mn);
      float p1 = __expf(e1 - mn);
      lrow[v] = lrow[v] * sc + half16_sum(p0 + p1);
      mrow[v] = mn;
      o0[v] *= sc;
      o1[v] *= sc;
      int row = 8 * hh + v;
      pbuf[row][nl]      = p0;
      pbuf[row][16 + nl] = p1;
    }
    __syncthreads();
    BF16x16 ap;                                   // P in A-fragment layout
#pragma unroll
    for (int e2 = 0; e2 < 8; ++e2) {
      ap.e[e2]     = (__bf16)pbuf[nl][hh * 16 + e2];
      ap.e[8 + e2] = (__bf16)pbuf[nl][hh * 16 + 8 + e2];
    }
    __syncthreads();
    o0 = wmma_bf16f32(ap.v, bv0, o0);
    o1 = wmma_bf16f32(ap.v, bv1, o1);
  }

#pragma unroll
  for (int v = 0; v < 8; ++v) {
    int row = 8 * hh + v;
    float inv = 1.0f / lrow[v];
    O[(size_t)(qb + row) * C_DIM + head * HEAD_DIM + nl]      = (__bf16)(o0[v] * inv);
    O[(size_t)(qb + row) * C_DIM + head * HEAD_DIM + 16 + nl] = (__bf16)(o1[v] * inv);
  }
}

// ---------------------------------------------------------------------------
extern "C" void kernel_launch(void* const* d_in, const int* in_sizes, int n_in,
                              void* d_out, int out_size, void* d_ws, size_t ws_size,
                              hipStream_t stream) {
  (void)in_sizes; (void)n_in; (void)out_size; (void)ws_size;
  const float* x     = (const float*)d_in[0];
  const float* gn1_w = (const float*)d_in[1];
  const float* gn1_b = (const float*)d_in[2];
  const float* wq = (const float*)d_in[3];  const float* bq = (const float*)d_in[4];
  const float* wk = (const float*)d_in[5];  const float* bk = (const float*)d_in[6];
  const float* wv = (const float*)d_in[7];  const float* bv = (const float*)d_in[8];
  const float* wo = (const float*)d_in[9];  const float* bo = (const float*)d_in[10];
  const float* gn2_w = (const float*)d_in[11];
  const float* gn2_b = (const float*)d_in[12];
  const float* w1 = (const float*)d_in[13]; const float* b1 = (const float*)d_in[14];
  const float* w2 = (const float*)d_in[15]; const float* b2 = (const float*)d_in[16];

  char* ws = (char*)d_ws;
  size_t off = 0;
  auto alloc = [&](size_t bytes) -> void* {
    void* p = ws + off;
    off += (bytes + 255) & ~(size_t)255;
    return p;
  };
  float*  t     = (float*) alloc((size_t)N_TOK * C_DIM * 4);   // gn1 out, token-major
  float*  x2    = (float*) alloc((size_t)N_TOK * C_DIM * 4);   // xn + attn
  float*  t2    = (float*) alloc((size_t)N_TOK * C_DIM * 4);   // gn2 out
  __bf16* t_bf  = (__bf16*)alloc((size_t)N_TOK * C_DIM * 2);
  __bf16* t2_bf = (__bf16*)alloc((size_t)N_TOK * C_DIM * 2);
  __bf16* q_bf  = (__bf16*)alloc((size_t)N_TOK * C_DIM * 2);
  __bf16* k_bf  = (__bf16*)alloc((size_t)N_TOK * C_DIM * 2);
  __bf16* v_t   = (__bf16*)alloc((size_t)C_DIM * N_TOK * 2);   // V^T (per-head rows)
  __bf16* a_o   = (__bf16*)alloc((size_t)N_TOK * C_DIM * 2);   // attention output
  __bf16* h_bf  = (__bf16*)alloc((size_t)N_TOK * FF_DIM * 2);  // FFN hidden
  __bf16* wq_t  = (__bf16*)alloc((size_t)C_DIM * C_DIM * 2);
  __bf16* wk_t  = (__bf16*)alloc((size_t)C_DIM * C_DIM * 2);
  __bf16* wv_t  = (__bf16*)alloc((size_t)C_DIM * C_DIM * 2);
  __bf16* wo_t  = (__bf16*)alloc((size_t)C_DIM * C_DIM * 2);
  __bf16* w1_t  = (__bf16*)alloc((size_t)C_DIM * FF_DIM * 2);
  __bf16* w2_t  = (__bf16*)alloc((size_t)FF_DIM * C_DIM * 2);

  const int cc = C_DIM * C_DIM, cf = C_DIM * FF_DIM;
  convert_transpose_kernel<<<(cc + 255) / 256, 256, 0, stream>>>(wq, wq_t, C_DIM, C_DIM);
  convert_transpose_kernel<<<(cc + 255) / 256, 256, 0, stream>>>(wk, wk_t, C_DIM, C_DIM);
  convert_transpose_kernel<<<(cc + 255) / 256, 256, 0, stream>>>(wv, wv_t, C_DIM, C_DIM);
  convert_transpose_kernel<<<(cc + 255) / 256, 256, 0, stream>>>(wo, wo_t, C_DIM, C_DIM);
  convert_transpose_kernel<<<(cf + 255) / 256, 256, 0, stream>>>(w1, w1_t, C_DIM, FF_DIM);
  convert_transpose_kernel<<<(cf + 255) / 256, 256, 0, stream>>>(w2, w2_t, FF_DIM, C_DIM);

  groupnorm1_kernel<<<32, 256, 0, stream>>>(x, gn1_w, gn1_b, t, t_bf);

  dim3 gCC(N_TOK / 64, C_DIM / 64);
  wmma_gemm_kernel<<<gCC, 128, 0, stream>>>(t_bf, wq_t, bq, nullptr, q_bf, N_TOK, C_DIM, C_DIM, GEMM_BIAS_BF16);
  wmma_gemm_kernel<<<gCC, 128, 0, stream>>>(t_bf, wk_t, bk, nullptr, k_bf, N_TOK, C_DIM, C_DIM, GEMM_BIAS_BF16);
  wmma_gemm_kernel<<<gCC, 128, 0, stream>>>(t_bf, wv_t, bv, nullptr, v_t,  N_TOK, C_DIM, C_DIM, GEMM_BIAS_BF16_T);

  flash_attn_kernel<<<dim3(N_TOK / 16, N_HEADS), 32, 0, stream>>>(q_bf, k_bf, v_t, a_o);

  wmma_gemm_kernel<<<gCC, 128, 0, stream>>>(a_o, wo_t, bo, t, x2, N_TOK, C_DIM, C_DIM, GEMM_BIAS_RES_F32);

  groupnorm2_kernel<<<32, 256, 0, stream>>>(x2, gn2_w, gn2_b, t2, t2_bf);

  wmma_gemm_kernel<<<dim3(N_TOK / 64, FF_DIM / 64), 128, 0, stream>>>(
      t2_bf, w1_t, b1, nullptr, h_bf, N_TOK, FF_DIM, C_DIM, GEMM_BIAS_GELU_BF16);
  wmma_gemm_kernel<<<gCC, 128, 0, stream>>>(
      h_bf, w2_t, b2, t2, (float*)d_out, N_TOK, C_DIM, FF_DIM, GEMM_BIAS_RES_F32_T);
}
